// PolicyNetwork_23596550324766
// MI455X (gfx1250) — matrix operand order repaired and checked by reading, not compiled, not run on hardware
//
#include <hip/hip_runtime.h>
#include <hip/hip_bf16.h>

#define N_NODES   10000
#define N_EDGES   160000
#define D_IN      128
#define HDIM      1024
#define NGRAPHS   8

typedef __attribute__((ext_vector_type(16))) __bf16 v16bf;
typedef __attribute__((ext_vector_type(8)))  float  v8f;

static __device__ __forceinline__ unsigned short f32_to_bf16_rne(float f) {
    unsigned int u = __float_as_uint(f);
    unsigned int r = (u + 0x7FFFu + ((u >> 16) & 1u)) >> 16;
    return (unsigned short)r;
}

// ---------------- small utility kernels ----------------

__global__ void zero_f32_kernel(float* p, int n) {
    int i = blockIdx.x * blockDim.x + threadIdx.x;
    if (i < n) p[i] = 0.0f;
}

__global__ void init_deg_kernel(float* deg, int n) {
    int i = blockIdx.x * blockDim.x + threadIdx.x;
    if (i < n) deg[i] = 1.0f;   // self-loop contribution
}

__global__ void deg_acc_kernel(const int* __restrict__ dst, float* deg, int e) {
    int i = blockIdx.x * blockDim.x + threadIdx.x;
    if (i < e) atomicAdd(&deg[dst[i]], 1.0f);
}

__global__ void dinv_kernel(const float* __restrict__ deg, float* dinv, int n) {
    int i = blockIdx.x * blockDim.x + threadIdx.x;
    if (i < n) dinv[i] = rsqrtf(fmaxf(deg[i], 1.0f));
}

__global__ void cvt_bf16_kernel(const float* __restrict__ in, unsigned short* __restrict__ out, int n) {
    int i = blockIdx.x * blockDim.x + threadIdx.x;
    if (i < n) out[i] = f32_to_bf16_rne(in[i]);
}

// ---------------- WMMA bf16 GEMM: C[M,Hn] = A[M,K] @ W[K,Hn] ----------------
// A, W16: bf16 row-major; C: fp32.
// Block = 256 thr (8 waves). Block tile: 256 rows x 64 cols. Per wave: 32 rows
// (2 A frags), 8 WMMAs per K-step of 32. Double-buffered fragment-ready B tile
// in LDS (padded stride 24 elems = 48B to spread banks, keeps 16B alignment).
// All 4 B fragments are loaded up-front each step so LDS latency overlaps WMMA.

#define FR_STRIDE 24                 // padded elems per fragment-lane row
#define FRAG_SZ   (32 * FR_STRIDE)   // 768 elems per 16x32 B fragment
#define TILE_ELEMS (4 * FRAG_SZ)     // 4 fragments per 32x64 tile

static __device__ __forceinline__ void store_bfrag(unsigned short* dp, uint4 wv) {
    dp[0 * FR_STRIDE] = (unsigned short)(wv.x & 0xffffu);
    dp[1 * FR_STRIDE] = (unsigned short)(wv.x >> 16);
    dp[2 * FR_STRIDE] = (unsigned short)(wv.y & 0xffffu);
    dp[3 * FR_STRIDE] = (unsigned short)(wv.y >> 16);
    dp[4 * FR_STRIDE] = (unsigned short)(wv.z & 0xffffu);
    dp[5 * FR_STRIDE] = (unsigned short)(wv.z >> 16);
    dp[6 * FR_STRIDE] = (unsigned short)(wv.w & 0xffffu);
    dp[7 * FR_STRIDE] = (unsigned short)(wv.w >> 16);
}

__global__ __launch_bounds__(256)
void gemm_bf16_wmma_kernel(const unsigned short* __restrict__ A,
                           const unsigned short* __restrict__ W16,
                           float* __restrict__ C,
                           int M, int K, int Hn)
{
    __shared__ __align__(16) unsigned short Bs[2][TILE_ELEMS];

    const int tid  = threadIdx.x;
    const int wave = tid >> 5;
    const int lane = tid & 31;
    const int half = lane >> 4;   // 0 or 1
    const int lrow = lane & 15;

    const int rowBase = blockIdx.x * 256 + wave * 32;
    const int colBase = blockIdx.y * 64;

    const int m0 = rowBase + lrow;
    const int m1 = m0 + 16;
    const bool mval0 = (m0 < M);
    const bool mval1 = (m1 < M);
    const unsigned short* Arow0 = A + (mval0 ? (size_t)m0 * K : 0);
    const unsigned short* Arow1 = A + (mval1 ? (size_t)m1 * K : 0);

    // Cooperative W loader: thread t loads 8 consecutive bf16 (one uint4) of
    // tile row lk at columns ln..ln+7, scatters to fragment-ready layout.
    const int lk = tid >> 3;           // 0..31  (k within tile)
    const int ln = (tid & 7) * 8;      // 0,8,...,56 (n within tile)
    const unsigned short* Wp = W16 + (size_t)lk * Hn + colBase + ln;
    const int fragBase = (ln >> 4) * FRAG_SZ
                       + (((lk >> 4) << 4) + (ln & 15)) * FR_STRIDE
                       + (lk & 15);

    v8f acc[8];
    v8f vz = {};
    #pragma unroll
    for (int i = 0; i < 8; i++) acc[i] = vz;

    // preload tile 0
    store_bfrag(&Bs[0][fragBase], *(const uint4*)Wp);
    __syncthreads();

    int buf = 0;
    for (int kt = 0; kt < K; kt += 32) {
        const int nxt = kt + 32;
        uint4 wnext = make_uint4(0u, 0u, 0u, 0u);
        if (nxt < K) wnext = *(const uint4*)(Wp + (size_t)nxt * Hn);

        // A fragments (16-bit 16x32 layout): two contiguous 16B runs per lane.
        union { v16bf v; uint4 q[2]; } a0, a1;
        if (mval0) {
            a0.q[0] = *(const uint4*)(Arow0 + kt + half * 16);
            a0.q[1] = *(const uint4*)(Arow0 + kt + half * 16 + 8);
        } else {
            uint4 z = make_uint4(0u, 0u, 0u, 0u); a0.q[0] = z; a0.q[1] = z;
        }
        if (mval1) {
            a1.q[0] = *(const uint4*)(Arow1 + kt + half * 16);
            a1.q[1] = *(const uint4*)(Arow1 + kt + half * 16 + 8);
        } else {
            uint4 z = make_uint4(0u, 0u, 0u, 0u); a1.q[0] = z; a1.q[1] = z;
        }

        // streaming-A prefetch two tiles ahead (global_prefetch_b8)
        if (kt + 64 < K) {
            __builtin_prefetch(Arow0 + kt + 64, 0, 1);
            __builtin_prefetch(Arow1 + kt + 64, 0, 1);
        }

        // Issue all 8 ds_load_b128 up-front so waits amortize across WMMAs.
        union { v16bf v; uint4 q[2]; } bfrag[4];
        #pragma unroll
        for (int j = 0; j < 4; j++) {
            const uint4* bp = (const uint4*)&Bs[buf][j * FRAG_SZ + lane * FR_STRIDE];
            bfrag[j].q[0] = bp[0];
            bfrag[j].q[1] = bp[1];
        }

        #pragma unroll
        for (int j = 0; j < 4; j++) {
            acc[j]     = __builtin_amdgcn_wmma_f32_16x16x32_bf16(
                false, a0.v, false, bfrag[j].v, (short)0, acc[j],     false, false);
            acc[4 + j] = __builtin_amdgcn_wmma_f32_16x16x32_bf16(
                false, a1.v, false, bfrag[j].v, (short)0, acc[4 + j], false, false);
        }

        if (nxt < K) {
            store_bfrag(&Bs[buf ^ 1][fragBase], wnext);
            __syncthreads();
            buf ^= 1;
        }
    }

    // C layout: lane -> n = lane%16, row = r + 8*(lane/16)
    if (rowBase + 32 <= M) {
        // fast path: whole 32-row wave tile in range, no per-element guards
        #pragma unroll
        for (int j = 0; j < 4; j++) {
            int n = colBase + j * 16 + lrow;
            #pragma unroll
            for (int r = 0; r < 8; r++) {
                int mmA = rowBase + half * 8 + r;
                C[(size_t)mmA * Hn + n]        = acc[j][r];
                C[(size_t)(mmA + 16) * Hn + n] = acc[4 + j][r];
            }
        }
    } else {
        #pragma unroll
        for (int j = 0; j < 4; j++) {
            int n = colBase + j * 16 + lrow;
            #pragma unroll
            for (int r = 0; r < 8; r++) {
                int mmA = rowBase + half * 8 + r;
                if (mmA < M)      C[(size_t)mmA * Hn + n] = acc[j][r];
                int mmB = mmA + 16;
                if (mmB < M)      C[(size_t)mmB * Hn + n] = acc[4 + j][r];
            }
        }
    }
}

// ---------------- scatter-add over edges (GCN aggregation) ----------------
// One block per edge (E real edges + N self loops). 256 thr x 4 cols (H=1024).

__global__ __launch_bounds__(256)
void scatter_kernel(const float* __restrict__ t,
                    const int* __restrict__ src, const int* __restrict__ dst,
                    const float* __restrict__ dinv,
                    float* __restrict__ agg, int E_, int Hn)
{
    int e = blockIdx.x;
    int s, d;
    if (e < E_) { s = src[e]; d = dst[e]; }
    else        { s = e - E_; d = s; }
    float norm = dinv[s] * dinv[d];

    int tid = threadIdx.x;
    const float4* tr = (const float4*)(t + (size_t)s * Hn);
    float4 val = tr[tid];
    size_t ab = (size_t)d * Hn + tid * 4;
    atomicAdd(&agg[ab + 0], val.x * norm);
    atomicAdd(&agg[ab + 1], val.y * norm);
    atomicAdd(&agg[ab + 2], val.z * norm);
    atomicAdd(&agg[ab + 3], val.w * norm);
}

// ---------------- LayerNorm + ReLU (+bias, +residual, optional pre-LN store) ----------------

__global__ __launch_bounds__(256)
void ln_relu_kernel(const float* __restrict__ agg, const float* __restrict__ bias,
                    const float* __restrict__ res,
                    const float* __restrict__ g, const float* __restrict__ b,
                    float* __restrict__ out_pre, float* __restrict__ out_h, int Hn)
{
    __shared__ float red[256];
    int row = blockIdx.x;
    int tid = threadIdx.x;
    size_t base = (size_t)row * Hn;

    float v[4];
    float s = 0.0f;
    #pragma unroll
    for (int j = 0; j < 4; j++) {
        int c = tid + j * 256;
        float x = agg[base + c] + bias[c];
        if (res) x += res[base + c];
        v[j] = x;
        s += x;
    }
    if (out_pre) {
        #pragma unroll
        for (int j = 0; j < 4; j++) out_pre[base + tid + j * 256] = v[j];
    }

    red[tid] = s; __syncthreads();
    for (int off = 128; off > 0; off >>= 1) {
        if (tid < off) red[tid] += red[tid + off];
        __syncthreads();
    }
    float mean = red[0] / (float)Hn;
    __syncthreads();

    float s2 = 0.0f;
    #pragma unroll
    for (int j = 0; j < 4; j++) { float dl = v[j] - mean; s2 += dl * dl; }
    red[tid] = s2; __syncthreads();
    for (int off = 128; off > 0; off >>= 1) {
        if (tid < off) red[tid] += red[tid + off];
        __syncthreads();
    }
    float rstd = rsqrtf(red[0] / (float)Hn + 1e-5f);

    #pragma unroll
    for (int j = 0; j < 4; j++) {
        int c = tid + j * 256;
        float y = (v[j] - mean) * rstd * g[c] + b[c];
        out_h[base + c] = fmaxf(y, 0.0f);
    }
}

// ---------------- global mean pool ----------------

__global__ __launch_bounds__(256)
void pool_acc_kernel(const float* __restrict__ emb, const int* __restrict__ batch,
                     float* __restrict__ sums, float* __restrict__ counts, int Hn)
{
    int row = blockIdx.x;
    int bg = batch[row];
    int tid = threadIdx.x;
    if (tid == 0) atomicAdd(&counts[bg], 1.0f);
    #pragma unroll
    for (int j = 0; j < 4; j++) {
        int c = tid + j * 256;
        atomicAdd(&sums[(size_t)bg * Hn + c], emb[(size_t)row * Hn + c]);
    }
}

__global__ void pool_div_kernel(const float* __restrict__ sums, const float* __restrict__ counts,
                                float* __restrict__ gout, int Hn)
{
    int i = blockIdx.x * blockDim.x + threadIdx.x; // NGRAPHS*Hn
    int g = i / Hn;
    gout[i] = sums[i] / fmaxf(counts[g], 1.0f);
}

// ---------------- tiny MLP head ----------------

__global__ __launch_bounds__(256)
void mlp_kernel(const float* __restrict__ vin, const float* __restrict__ W,
                const float* __restrict__ bias, float* __restrict__ vout, int Hn)
{
    int gid = blockIdx.x * blockDim.x + threadIdx.x; // NGRAPHS*Hn
    int row = gid / Hn, col = gid % Hn;
    const float* vr = vin + (size_t)row * Hn;
    float acc = bias[col];
    for (int k = 0; k < Hn; k++) acc = fmaf(vr[k], W[(size_t)k * Hn + col], acc);
    vout[gid] = fmaxf(acc, 0.0f);
}

__global__ __launch_bounds__(256)
void value_kernel(const float* __restrict__ v, const float* __restrict__ vWo,
                  const float* __restrict__ vbo, float* __restrict__ out, int Hn)
{
    __shared__ float red[256];
    int gidx = blockIdx.x;
    int tid = threadIdx.x;
    float s = 0.0f;
    #pragma unroll
    for (int j = 0; j < 4; j++) {
        int k = tid + j * 256;
        s = fmaf(v[(size_t)gidx * Hn + k], vWo[k], s);
    }
    red[tid] = s; __syncthreads();
    for (int off = 128; off > 0; off >>= 1) {
        if (tid < off) red[tid] += red[tid + off];
        __syncthreads();
    }
    if (tid == 0) out[gidx] = red[0] + vbo[0];
}

// ---------------- host launch ----------------

extern "C" void kernel_launch(void* const* d_in, const int* in_sizes, int n_in,
                              void* d_out, int out_size, void* d_ws, size_t ws_size,
                              hipStream_t stream)
{
    (void)in_sizes; (void)n_in; (void)out_size; (void)ws_size;

    const float* x     = (const float*)d_in[0];
    const int*   ei    = (const int*)d_in[1];
    const int*   batch = (const int*)d_in[2];
    const float* W1    = (const float*)d_in[3];
    const float* b1    = (const float*)d_in[4];
    const float* Wh    = (const float*)d_in[5];
    const float* bh    = (const float*)d_in[6];
    const float* ln_g  = (const float*)d_in[7];
    const float* ln_b  = (const float*)d_in[8];
    const float* vW    = (const float*)d_in[9];
    const float* vb    = (const float*)d_in[10];
    const float* vWo   = (const float*)d_in[11];
    const float* vbo   = (const float*)d_in[12];

    const int* src = ei;
    const int* dst = ei + N_EDGES;

    // workspace bump allocator (256B aligned)
    size_t off = 0;
    auto alloc = [&](size_t bytes) -> void* {
        void* p = (char*)d_ws + off;
        off += (bytes + 255) & ~(size_t)255;
        return p;
    };
    float* deg  = (float*)alloc((size_t)N_NODES * 4);
    float* dinv = (float*)alloc((size_t)N_NODES * 4);
    float* t    = (float*)alloc((size_t)N_NODES * HDIM * 4);
    float* agg  = (float*)alloc((size_t)N_NODES * HDIM * 4);
    float* x1   = (float*)alloc((size_t)N_NODES * HDIM * 4);
    float* h    = (float*)alloc((size_t)N_NODES * HDIM * 4);
    unsigned short* A16 = (unsigned short*)alloc((size_t)N_NODES * HDIM * 2);
    unsigned short* W16 = (unsigned short*)alloc((size_t)HDIM * HDIM * 2);
    float* sums   = (float*)alloc((size_t)NGRAPHS * HDIM * 4);
    float* counts = (float*)alloc((size_t)NGRAPHS * 4);
    float* v0     = (float*)alloc((size_t)NGRAPHS * HDIM * 4);
    float* v1     = (float*)alloc((size_t)NGRAPHS * HDIM * 4);

    float* node_emb  = (float*)d_out;                               // [N, H]
    float* graph_emb = node_emb + (size_t)N_NODES * HDIM;           // [8, H]
    float* sval      = graph_emb + (size_t)NGRAPHS * HDIM;          // [8]

    const int NH = N_NODES * HDIM;
    dim3 gemmGrid((N_NODES + 255) / 256, HDIM / 64);

    // 1) degrees + symmetric norm
    init_deg_kernel<<<(N_NODES + 255) / 256, 256, 0, stream>>>(deg, N_NODES);
    deg_acc_kernel<<<(N_EDGES + 255) / 256, 256, 0, stream>>>(dst, deg, N_EDGES);
    dinv_kernel<<<(N_NODES + 255) / 256, 256, 0, stream>>>(deg, dinv, N_NODES);

    // 2) layer 1: x @ W1 -> scatter -> x1 = agg+b1 ; h = relu(LN(x1))
    cvt_bf16_kernel<<<(N_NODES * D_IN + 255) / 256, 256, 0, stream>>>(x, A16, N_NODES * D_IN);
    cvt_bf16_kernel<<<(D_IN * HDIM + 255) / 256, 256, 0, stream>>>(W1, W16, D_IN * HDIM);
    gemm_bf16_wmma_kernel<<<gemmGrid, 256, 0, stream>>>(A16, W16, t, N_NODES, D_IN, HDIM);
    zero_f32_kernel<<<(NH + 255) / 256, 256, 0, stream>>>(agg, NH);
    scatter_kernel<<<N_EDGES + N_NODES, 256, 0, stream>>>(t, src, dst, dinv, agg, N_EDGES, HDIM);
    ln_relu_kernel<<<N_NODES, 256, 0, stream>>>(agg, b1, nullptr, ln_g, ln_b, x1, h, HDIM);

    // 3) layers 2..3: gcn(h, Wh[L]) ; h = relu(LN(xL + x1))
    for (int L = 0; L < 2; L++) {
        cvt_bf16_kernel<<<(NH + 255) / 256, 256, 0, stream>>>(h, A16, NH);
        cvt_bf16_kernel<<<(HDIM * HDIM + 255) / 256, 256, 0, stream>>>(
            Wh + (size_t)L * HDIM * HDIM, W16, HDIM * HDIM);
        gemm_bf16_wmma_kernel<<<gemmGrid, 256, 0, stream>>>(A16, W16, t, N_NODES, HDIM, HDIM);
        zero_f32_kernel<<<(NH + 255) / 256, 256, 0, stream>>>(agg, NH);
        scatter_kernel<<<N_EDGES + N_NODES, 256, 0, stream>>>(t, src, dst, dinv, agg, N_EDGES, HDIM);
        ln_relu_kernel<<<N_NODES, 256, 0, stream>>>(agg, bh + (size_t)L * HDIM, x1,
                                                    ln_g + (size_t)(L + 1) * HDIM,
                                                    ln_b + (size_t)(L + 1) * HDIM,
                                                    nullptr, h, HDIM);
    }

    // 4) layer 4: gcn(h, Wh[2]) ; node_emb = relu(LN(x4 + h))
    cvt_bf16_kernel<<<(NH + 255) / 256, 256, 0, stream>>>(h, A16, NH);
    cvt_bf16_kernel<<<(HDIM * HDIM + 255) / 256, 256, 0, stream>>>(
        Wh + (size_t)2 * HDIM * HDIM, W16, HDIM * HDIM);
    gemm_bf16_wmma_kernel<<<gemmGrid, 256, 0, stream>>>(A16, W16, t, N_NODES, HDIM, HDIM);
    zero_f32_kernel<<<(NH + 255) / 256, 256, 0, stream>>>(agg, NH);
    scatter_kernel<<<N_EDGES + N_NODES, 256, 0, stream>>>(t, src, dst, dinv, agg, N_EDGES, HDIM);
    ln_relu_kernel<<<N_NODES, 256, 0, stream>>>(agg, bh + (size_t)2 * HDIM, h,
                                                ln_g + (size_t)3 * HDIM, ln_b + (size_t)3 * HDIM,
                                                nullptr, node_emb, HDIM);

    // 5) global mean pool
    zero_f32_kernel<<<(NGRAPHS * HDIM + NGRAPHS + 255) / 256, 256, 0, stream>>>(
        sums, NGRAPHS * HDIM + NGRAPHS);
    pool_acc_kernel<<<N_NODES, 256, 0, stream>>>(node_emb, batch, sums, counts, HDIM);
    pool_div_kernel<<<(NGRAPHS * HDIM + 255) / 256, 256, 0, stream>>>(sums, counts, graph_emb, HDIM);

    // 6) MLP head (6 layers, ping-pong) + scalar value
    const float* cur = graph_emb;
    float* bufs[2] = { v0, v1 };
    for (int i = 0; i < 6; i++) {
        float* outb = bufs[i & 1];
        mlp_kernel<<<(NGRAPHS * HDIM + 255) / 256, 256, 0, stream>>>(
            cur, vW + (size_t)i * HDIM * HDIM, vb + (size_t)i * HDIM, outb, HDIM);
        cur = outb;
    }
    value_kernel<<<NGRAPHS, 256, 0, stream>>>(cur, vWo, vbo, sval, HDIM);
}
